// STKAttentionV5_584115552626
// MI455X (gfx1250) — matrix operand order repaired and checked by reading, not compiled, not run on hardware
//
#include <hip/hip_runtime.h>
#include <hip/hip_bf16.h>

// ---------------------------------------------------------------------------
// STK attention block for gfx1250 (MI455X).
//  - Tensors kept fp32 in memory (bfhwc layout, b=2 f=16 h=64 w=64 c=16).
//  - 16x16 channel projections run on v_wmma_f32_16x16x32_f16 (K padded to 32).
//    B (weight) fragments are loop-invariant: built once per wave from
//    unconditional coalesced loads (no predicated-load chains).
//  - LDS activations for the A operand are kept in f16 so the A fragment is a
//    single aligned ds_load_b128 per lane.
//  - Scores/softmax (head_dim=2) run on VALU with online softmax.
//  - Convs are LDS-tiled VALU loops (small: 144 MACs per output).
// Workspace: X (NT floats) | R (NT floats) | e (32 floats)
// ---------------------------------------------------------------------------

typedef __attribute__((ext_vector_type(16))) _Float16 v16h;
typedef __attribute__((ext_vector_type(8)))  _Float16 v8h;
typedef __attribute__((ext_vector_type(8)))  float    v8f;

#define NT   2097152   // 2*16*64*64*16 elements per tensor
#define MIDC 16

// ---- WMMA fragment helpers (layouts per CDNA5 ISA 7.12.2) ------------------

// A: 16x32 f16, rows = tokens, K = channels (0..15 real, 16..31 zero pad).
// lane<16: element i<8 -> K=i ; lane>=16: element i<8 -> K=i+8 ; i>=8 -> zero.
// T is an f16 LDS array [rows][16]; the 8 needed halves are 16B-aligned.
__device__ inline v16h load_A_frag(const _Float16* T, int tilebase, int lane) {
  const int row = tilebase + (lane & 15);
  const int kb  = (lane < 16) ? 0 : 8;
  const v8h lo = *(const v8h*)(T + row * MIDC + kb);
  v16h a;
#pragma unroll
  for (int i = 0; i < 8; ++i) a[i] = lo[i];
#pragma unroll
  for (int i = 8; i < 16; ++i) a[i] = (_Float16)0.0f;
  return a;
}

// B: 32x16 f16, B[k][n] = W[n][k] so that D = X @ W^T.
// lane<16: element i<8 -> K=i ; lane>=16: element i<8 -> K=i+8 ; i>=8 -> zero pad.
// Loads are unconditional (every lane reads a valid row); zeroing is a value
// select so no EXEC-predicated load chains are generated.
__device__ inline v16h load_B_frag(const float* __restrict__ W, int lane) {
  const int n = lane & 15;
  const float4* wp = (const float4*)(W + n * 16);
  const float4 w0 = wp[0], w1 = wp[1], w2 = wp[2], w3 = wp[3];
  const float t[16] = {w0.x, w0.y, w0.z, w0.w, w1.x, w1.y, w1.z, w1.w,
                       w2.x, w2.y, w2.z, w2.w, w3.x, w3.y, w3.z, w3.w};
  const bool hi = (lane >= 16);
  v16h b;
#pragma unroll
  for (int i = 0; i < 8; ++i) b[i] = (_Float16)(hi ? t[i + 8] : t[i]);
#pragma unroll
  for (int i = 8; i < 16; ++i) b[i] = (_Float16)0.0f;
  return b;
}

// D/C: f32 16x16; VGPR r, lane L -> (M = r + (L>=16?8:0), N = L%16)
__device__ inline void store_D(float* T, int tilebase, int lane, v8f d) {
  const int n  = lane & 15;
  const int mb = tilebase + ((lane < 16) ? 0 : 8);
#pragma unroll
  for (int r = 0; r < 8; ++r) T[(mb + r) * MIDC + n] = d[r];
}

// ---- timestep embedding: e = silu(emb @ W1^T + b1) @ W2^T + b2 -------------
__global__ void stk_emb_kernel(const float* __restrict__ emb,
                               const float* __restrict__ w1, const float* __restrict__ b1,
                               const float* __restrict__ w2, const float* __restrict__ b2,
                               float* __restrict__ e) {
  __shared__ float h1[32];
  const int t = threadIdx.x;          // 32 threads: (b, j)
  const int b = t >> 4, j = t & 15;
  float acc = b1[j];
  for (int k = 0; k < 512; ++k) acc += emb[b * 512 + k] * w1[j * 512 + k];
  acc = acc / (1.0f + __expf(-acc));  // silu
  h1[t] = acc;
  __syncthreads();
  float acc2 = b2[j];
#pragma unroll
  for (int k = 0; k < 16; ++k) acc2 += h1[b * 16 + k] * w2[j * 16 + k];
  e[t] = acc2;
}

// ---- conv stem: X = conv(hidden), R = conv(ref) + e ------------------------
__global__ __launch_bounds__(256) void stk_conv_in_kernel(
    const float* __restrict__ hidden, const float* __restrict__ refs,
    const float* __restrict__ w, const float* __restrict__ bias,
    const float* __restrict__ e, float* __restrict__ X, float* __restrict__ Rr) {
  __shared__ float wl[2304];          // [co][ci][ky][kx]
  __shared__ float rows[3072];        // [3][64][16]
  const int t    = threadIdx.x;
  const int bid  = blockIdx.x;        // 0..4095
  const int tsel = bid >> 11;         // 0 = hidden, 1 = ref
  const int rem  = bid & 2047;
  const int bf   = rem >> 6;
  const int y    = rem & 63;
  const float* src = tsel ? refs : hidden;
  for (int i = t; i < 2304; i += 256) wl[i] = w[i];
  for (int i = t; i < 3072; i += 256) {
    const int dy = i >> 10;
    const int r2 = i & 1023;
    const int yy = y + dy - 1;
    rows[i] = (yy >= 0 && yy < 64) ? src[(bf * 64 + yy) * 1024 + r2] : 0.0f;
  }
  __syncthreads();
  float* dst = tsel ? Rr : X;
  const int b = bf >> 4;
#pragma unroll
  for (int j = 0; j < 4; ++j) {
    const int oi = t + j * 256;       // x*16 + co
    const int x = oi >> 4, co = oi & 15;
    float acc = bias[co];
#pragma unroll
    for (int ky = 0; ky < 3; ++ky) {
#pragma unroll
      for (int kx = 0; kx < 3; ++kx) {
        const int xx = x + kx - 1;
        if (xx < 0 || xx >= 64) continue;
        const float* rp = &rows[(ky * 64 + xx) * 16];
        const float* wp = &wl[co * 144 + ky * 3 + kx];
#pragma unroll
        for (int ci = 0; ci < 16; ++ci) acc += rp[ci] * wp[ci * 9];
      }
    }
    if (tsel) acc += e[b * 16 + co];
    dst[(bf * 64 + y) * 1024 + oi] = acc;
  }
}

// ---- axis attention: mode 0=row (S=128), 1=col (S=128), 2=temporal (S=32) --
__global__ __launch_bounds__(128) void stk_attn_kernel(
    float* __restrict__ X, const float* __restrict__ R,
    const float* __restrict__ Wq, const float* __restrict__ Wk,
    const float* __restrict__ Wv, const float* __restrict__ Wo,
    const float* __restrict__ bo,
    const float* __restrict__ ng, const float* __restrict__ nb,
    int mode) {
  __shared__ _Float16 tokh[128 * MIDC];   // normalized tokens (A operand), f16
  __shared__ float    qs  [128 * MIDC];
  __shared__ float    ks  [128 * MIDC];
  __shared__ float    vs  [128 * MIDC];
  __shared__ _Float16 osh [128 * MIDC];   // attention output (A operand), f16

  const int tid  = threadIdx.x;
  const int lane = tid & 31;
  const int wave = tid >> 5;
  const int bid  = blockIdx.x;

  int S, halfS, base, stride;
  if (mode == 0) {                    // row: seq = (bf, y), token = x
    const int bf = bid >> 6, y = bid & 63;
    S = 128; halfS = 64; base = (bf * 64 + y) * 1024; stride = 16;
  } else if (mode == 1) {             // col: seq = (bf, x), token = y
    const int bf = bid >> 6, x = bid & 63;
    S = 128; halfS = 64; base = bf * 65536 + x * 16; stride = 1024;
  } else {                            // temporal: seq = (b, y, x), token = f
    const int b = bid >> 12, y = (bid >> 6) & 63, x = bid & 63;
    S = 32; halfS = 16; base = b * 1048576 + y * 1024 + x * 16; stride = 65536;
  }

  // Loop-invariant weight (B) fragments: built once per wave.
  const v16h bq = load_B_frag(Wq, lane);
  const v16h bk = load_B_frag(Wk, lane);
  const v16h bv = load_B_frag(Wv, lane);
  const v16h bw = load_B_frag(Wo, lane);

  // gather + posembed + layernorm (one thread per token)
  if (tid < S) {
    const float* src = (tid < halfS) ? (X + base + tid * stride)
                                     : (R + base + (tid - halfS) * stride);
    float v[16];
#pragma unroll
    for (int c = 0; c < 16; ++c) v[c] = src[c];
    // sincos 1D posembed, dim 16; omega_j = 10000^(-j/8)
    const float omega[8] = {1.0f, 0.31622776601683794f, 0.1f,
                            0.03162277660168379f, 0.01f,
                            0.0031622776601683794f, 0.001f,
                            0.00031622776601683794f};
#pragma unroll
    for (int j = 0; j < 8; ++j) {
      const float ag = (float)tid * omega[j];
      v[j]     += __sinf(ag);
      v[j + 8] += __cosf(ag);
    }
    float mean = 0.0f;
#pragma unroll
    for (int c = 0; c < 16; ++c) mean += v[c];
    mean *= 0.0625f;
    float var = 0.0f;
#pragma unroll
    for (int c = 0; c < 16; ++c) { const float d = v[c] - mean; var += d * d; }
    var *= 0.0625f;
    const float inv = rsqrtf(var + 1e-5f);
#pragma unroll
    for (int c = 0; c < 16; ++c)
      tokh[tid * MIDC + c] = (_Float16)((v[c] - mean) * inv * ng[c] + nb[c]);
  }
  __syncthreads();

  // QKV projections: one WMMA tile = 16 tokens; waves stripe over tiles
  const int ntiles = S >> 4;
  for (int tile = wave; tile < ntiles; tile += 4) {
    const int tb = tile << 4;
    const v16h a = load_A_frag(tokh, tb, lane);
    const v8f  z = {};
    v8f dq = __builtin_amdgcn_wmma_f32_16x16x32_f16(false, a, false, bq, (short)0, z, false, false);
    store_D(qs, tb, lane, dq);
    v8f dk = __builtin_amdgcn_wmma_f32_16x16x32_f16(false, a, false, bk, (short)0, z, false, false);
    store_D(ks, tb, lane, dk);
    v8f dv = __builtin_amdgcn_wmma_f32_16x16x32_f16(false, a, false, bv, (short)0, z, false, false);
    store_D(vs, tb, lane, dv);
  }
  __syncthreads();

  // online-softmax attention, head_dim = 2, 8 heads; one thread per token
  if (tid < S) {
    const int s = tid;
#pragma unroll
    for (int hd = 0; hd < 8; ++hd) {
      const float q0 = qs[s * MIDC + 2 * hd];
      const float q1 = qs[s * MIDC + 2 * hd + 1];
      float m = -1e30f, l = 0.0f, a0 = 0.0f, a1 = 0.0f;
      for (int t2 = 0; t2 < S; ++t2) {
        const float sc = (q0 * ks[t2 * MIDC + 2 * hd] +
                          q1 * ks[t2 * MIDC + 2 * hd + 1]) * 0.70710678f;
        const float nm   = fmaxf(m, sc);
        const float corr = __expf(m - nm);
        const float p    = __expf(sc - nm);
        l  = l * corr + p;
        a0 = a0 * corr + p * vs[t2 * MIDC + 2 * hd];
        a1 = a1 * corr + p * vs[t2 * MIDC + 2 * hd + 1];
        m = nm;
      }
      const float invl = 1.0f / l;
      osh[s * MIDC + 2 * hd]     = (_Float16)(a0 * invl);
      osh[s * MIDC + 2 * hd + 1] = (_Float16)(a1 * invl);
    }
  }
  __syncthreads();

  // output projection (+bias) for the first halfS tokens, written in place
  for (int tile = wave; tile < (halfS >> 4); tile += 4) {
    const int tb = tile << 4;
    const v16h a = load_A_frag(osh, tb, lane);
    const int n = lane & 15;
    v8f c;
    const float bia = bo[n];
#pragma unroll
    for (int r = 0; r < 8; ++r) c[r] = bia;
    v8f d = __builtin_amdgcn_wmma_f32_16x16x32_f16(false, a, false, bw, (short)0, c, false, false);
    const int mb = tb + ((lane < 16) ? 0 : 8);
#pragma unroll
    for (int r = 0; r < 8; ++r)
      X[base + (mb + r) * stride + n] = d[r];
  }
}

// ---- output conv + residual ------------------------------------------------
__global__ __launch_bounds__(256) void stk_conv_out_kernel(
    const float* __restrict__ Xin, const float* __restrict__ w,
    const float* __restrict__ bias, const float* __restrict__ hidden,
    float* __restrict__ out) {
  __shared__ float wl[2304];
  __shared__ float rows[3072];
  const int t   = threadIdx.x;
  const int bid = blockIdx.x;         // 0..2047
  const int bf  = bid >> 6;
  const int y   = bid & 63;
  for (int i = t; i < 2304; i += 256) wl[i] = w[i];
  for (int i = t; i < 3072; i += 256) {
    const int dy = i >> 10;
    const int r2 = i & 1023;
    const int yy = y + dy - 1;
    rows[i] = (yy >= 0 && yy < 64) ? Xin[(bf * 64 + yy) * 1024 + r2] : 0.0f;
  }
  __syncthreads();
#pragma unroll
  for (int j = 0; j < 4; ++j) {
    const int oi = t + j * 256;
    const int x = oi >> 4, co = oi & 15;
    float acc = bias[co];
#pragma unroll
    for (int ky = 0; ky < 3; ++ky) {
#pragma unroll
      for (int kx = 0; kx < 3; ++kx) {
        const int xx = x + kx - 1;
        if (xx < 0 || xx >= 64) continue;
        const float* rp = &rows[(ky * 64 + xx) * 16];
        const float* wp = &wl[co * 144 + ky * 3 + kx];
#pragma unroll
        for (int ci = 0; ci < 16; ++ci) acc += rp[ci] * wp[ci * 9];
      }
    }
    const int idx = (bf * 64 + y) * 1024 + oi;
    out[idx] = hidden[idx] + acc;
  }
}

// ---------------------------------------------------------------------------
extern "C" void kernel_launch(void* const* d_in, const int* in_sizes, int n_in,
                              void* d_out, int out_size, void* d_ws, size_t ws_size,
                              hipStream_t stream) {
  const float* hidden    = (const float*)d_in[0];
  const float* refs      = (const float*)d_in[1];
  const float* emb       = (const float*)d_in[2];
  const float* conv_in_w = (const float*)d_in[3];
  const float* conv_in_b = (const float*)d_in[4];
  const float* emb1_w    = (const float*)d_in[5];
  const float* emb1_b    = (const float*)d_in[6];
  const float* emb2_w    = (const float*)d_in[7];
  const float* emb2_b    = (const float*)d_in[8];
  const float* wqx = (const float*)d_in[9];
  const float* wkx = (const float*)d_in[10];
  const float* wvx = (const float*)d_in[11];
  const float* wox = (const float*)d_in[12];
  const float* box = (const float*)d_in[13];
  const float* wqy = (const float*)d_in[14];
  const float* wky = (const float*)d_in[15];
  const float* wvy = (const float*)d_in[16];
  const float* woy = (const float*)d_in[17];
  const float* boy = (const float*)d_in[18];
  const float* wqt = (const float*)d_in[19];
  const float* wkt = (const float*)d_in[20];
  const float* wvt = (const float*)d_in[21];
  const float* wot = (const float*)d_in[22];
  const float* bot = (const float*)d_in[23];
  const float* norm_g = (const float*)d_in[24];
  const float* norm_b = (const float*)d_in[25];
  const float* proj_w = (const float*)d_in[26];
  const float* proj_b = (const float*)d_in[27];

  float* ws = (float*)d_ws;
  float* X = ws;            // conv(hidden) -> resx -> resy -> rest (in place)
  float* R = ws + NT;       // conv(ref) + e
  float* E = ws + 2 * NT;   // timestep embedding [2][16]

  stk_emb_kernel<<<1, 32, 0, stream>>>(emb, emb1_w, emb1_b, emb2_w, emb2_b, E);
  stk_conv_in_kernel<<<4096, 256, 0, stream>>>(hidden, refs, conv_in_w, conv_in_b, E, X, R);
  stk_attn_kernel<<<2048, 128, 0, stream>>>(X, R, wqx, wkx, wvx, wox, box, norm_g, norm_b, 0);
  stk_attn_kernel<<<2048, 128, 0, stream>>>(X, R, wqy, wky, wvy, woy, boy, norm_g, norm_b, 1);
  stk_attn_kernel<<<8192, 128, 0, stream>>>(X, R, wqt, wkt, wvt, wot, bot, norm_g, norm_b, 2);
  stk_conv_out_kernel<<<2048, 256, 0, stream>>>(X, proj_w, proj_b, hidden, (float*)d_out);
}
